// SE3SequenceEncoder_82970178224147
// MI455X (gfx1250) — matrix operand order repaired and checked by reading, not compiled, not run on hardware
//
#include <hip/hip_runtime.h>
#include <math.h>

// ---------------------------------------------------------------------------
// Problem constants (from reference): B=2, N=384, D=256, Pd=64, L=8, H=8, Dh=32
// ---------------------------------------------------------------------------
#define BB   2
#define NN   384
#define DD   256
#define PD   64
#define LL   8
#define HH   8
#define DH   32
#define FF   1024
#define MM   (BB * NN)          // 768 rows

typedef __attribute__((ext_vector_type(16))) __bf16 v16bf;
typedef __attribute__((ext_vector_type(8)))  float  v8f;

__device__ __forceinline__ v8f wmma_bf16(v16bf a, v16bf b, v8f c) {
    // D = A(16x32 bf16) * B(32x16 bf16) + C(16x16 f32)
    return __builtin_amdgcn_wmma_f32_16x16x32_bf16(
        false, a, false, b, (short)0, c, false, false);
}

// Split an fp32 fragment into hi/lo bf16 fragments (bf16x3 scheme).
__device__ __forceinline__ void cvt_split(const float* f, v16bf& hi, v16bf& lo) {
#pragma unroll
    for (int i = 0; i < 16; ++i) {
        __bf16 h = (__bf16)f[i];
        hi[i] = h;
        lo[i] = (__bf16)(f[i] - (float)h);
    }
}

// 3-WMMA split accumulation: ~fp32 accuracy on the bf16 WMMA pipe.
__device__ __forceinline__ v8f wmma_split3(v16bf ah, v16bf al, v16bf bh, v16bf bl, v8f acc) {
    acc = wmma_bf16(al, bh, acc);
    acc = wmma_bf16(ah, bl, acc);
    acc = wmma_bf16(ah, bh, acc);
    return acc;
}

// fp32 A-fragment loader (attention): lane<16 -> K {0..15}, lane>=16 -> K {16..31}
__device__ __forceinline__ void load_a_frag(const float* rowp, int lane, float* out) {
    int koff = (lane < 16) ? 0 : 16;
#pragma unroll
    for (int j = 0; j < 8; ++j) out[j]     = rowp[koff + j];
#pragma unroll
    for (int j = 0; j < 8; ++j) out[8 + j] = rowp[koff + 8 + j];
}

// fp32 B-fragment loader (attention): element j at p[(kbase+j)*stride]
__device__ __forceinline__ void load_b_frag(const float* p, int stride, int lane, float* out) {
    int kbase = (lane < 16) ? 0 : 16;
#pragma unroll
    for (int j = 0; j < 16; ++j) out[j] = p[(kbase + j) * stride];
}

// Pre-converted bf16 A-fragment: one contiguous 16-element (32B) run.
__device__ __forceinline__ void load_a_frag_bf(const __bf16* rowp, int lane, v16bf& out) {
    int koff = (lane < 16) ? 0 : 16;
#pragma unroll
    for (int j = 0; j < 8; ++j) out[j]     = rowp[koff + j];
#pragma unroll
    for (int j = 0; j < 8; ++j) out[8 + j] = rowp[koff + 8 + j];
}

// Pre-converted, transposed bf16 B-fragment: one contiguous 16-element (32B) run.
__device__ __forceinline__ void load_b_frag_bf(const __bf16* p, int lane, v16bf& out) {
    int kbase = (lane < 16) ? 0 : 16;
#pragma unroll
    for (int j = 0; j < 16; ++j) out[j] = p[kbase + j];
}

__device__ __forceinline__ float wave_sum(float v) {
#pragma unroll
    for (int off = 16; off > 0; off >>= 1) v += __shfl_xor(v, off, 32);
    return v;
}
__device__ __forceinline__ float wave_max(float v) {
#pragma unroll
    for (int off = 16; off > 0; off >>= 1) v = fmaxf(v, __shfl_xor(v, off, 32));
    return v;
}

// ---------------------------------------------------------------------------
// Operand pre-conversion kernels (one conversion per element per GEMM use).
// ---------------------------------------------------------------------------
__global__ void cvt_act_kernel(const float* __restrict__ in,
                               __bf16* __restrict__ hi, __bf16* __restrict__ lo,
                               int n) {
    int i = blockIdx.x * blockDim.x + threadIdx.x;
    if (i >= n) return;
    float f = in[i];
    __bf16 h = (__bf16)f;
    hi[i] = h;
    lo[i] = (__bf16)(f - (float)h);
}

// W[K][N] fp32  ->  Wt[N][K] hi/lo bf16 (transposed so B-frag loads are contiguous)
__global__ void cvt_wt_kernel(const float* __restrict__ W,
                              __bf16* __restrict__ th, __bf16* __restrict__ tl,
                              int K, int N) {
    int i = blockIdx.x * blockDim.x + threadIdx.x;
    if (i >= K * N) return;
    int k = i / N, n = i % N;
    float f = W[i];
    __bf16 h = (__bf16)f;
    th[(size_t)n * K + k] = h;
    tl[(size_t)n * K + k] = (__bf16)(f - (float)h);
}

// ---------------------------------------------------------------------------
// WMMA GEMM on pre-converted operands: C[M,N] = epi(A[M,K] @ W[K,N])
// A given as row-major bf16 hi/lo; W given as TRANSPOSED [N][K] bf16 hi/lo.
// Each wave computes a 64x16 strip (4 m-subtiles share one B fragment per
// k-step -> 12 v_wmma per k-step per wave). Block = 128 threads, 64x64 tile.
// epilogue: +bias (optional) -> activation -> +resid (optional)
// ACT: 0 none, 1 exact GELU, 2 ReLU.
// ---------------------------------------------------------------------------
template <int ACT, bool RESID>
__global__ void gemm_wmma_kernel(const __bf16* __restrict__ Ah,
                                 const __bf16* __restrict__ Al,
                                 const __bf16* __restrict__ Wth,
                                 const __bf16* __restrict__ Wtl,
                                 const float* __restrict__ bias,
                                 const float* __restrict__ resid,
                                 float* __restrict__ C,
                                 int M, int K, int N) {
    int lane = threadIdx.x & 31;
    int wave = threadIdx.x >> 5;
    int tm = blockIdx.x * 64;                 // 4 m-subtiles
    int tn = blockIdx.y * 64 + wave * 16;     // one n-tile per wave
    if (tm >= M || tn >= N) return;

    int m = lane & 15;
    int n = lane & 15;
    const __bf16* bh_p = Wth + (size_t)(tn + n) * K;
    const __bf16* bl_p = Wtl + (size_t)(tn + n) * K;

    v8f acc0 = {}, acc1 = {}, acc2 = {}, acc3 = {};

    for (int k0 = 0; k0 < K; k0 += 32) {
        v16bf bh, bl;
        load_b_frag_bf(bh_p + k0, lane, bh);
        load_b_frag_bf(bl_p + k0, lane, bl);

        const __bf16* ahb = Ah + (size_t)(tm + m) * K + k0;
        const __bf16* alb = Al + (size_t)(tm + m) * K + k0;
        if (k0 + 32 < K) __builtin_prefetch(ahb + 32, 0, 0);

        v16bf ah, al;
        load_a_frag_bf(ahb, lane, ah);
        load_a_frag_bf(alb, lane, al);
        acc0 = wmma_split3(ah, al, bh, bl, acc0);

        load_a_frag_bf(ahb + 16 * K, lane, ah);
        load_a_frag_bf(alb + 16 * K, lane, al);
        acc1 = wmma_split3(ah, al, bh, bl, acc1);

        load_a_frag_bf(ahb + 32 * K, lane, ah);
        load_a_frag_bf(alb + 32 * K, lane, al);
        acc2 = wmma_split3(ah, al, bh, bl, acc2);

        load_a_frag_bf(ahb + 48 * K, lane, ah);
        load_a_frag_bf(alb + 48 * K, lane, al);
        acc3 = wmma_split3(ah, al, bh, bl, acc3);
    }

    float bv = bias ? bias[tn + n] : 0.f;
#pragma unroll
    for (int s = 0; s < 4; ++s) {
        v8f acc = (s == 0) ? acc0 : (s == 1) ? acc1 : (s == 2) ? acc2 : acc3;
#pragma unroll
        for (int r = 0; r < 8; ++r) {
            int row = tm + s * 16 + r + ((lane < 16) ? 0 : 8);
            float v = acc[r] + bv;
            if (ACT == 1) v = 0.5f * v * (1.f + erff(v * 0.70710678118654752f)); // exact GELU
            else if (ACT == 2) v = fmaxf(v, 0.f);
            if (RESID) v += resid[(size_t)row * N + tn + n];
            C[(size_t)row * N + tn + n] = v;
        }
    }
}

// ---------------------------------------------------------------------------
// SE(3) frame attention: scores = QK^T/sqrt(Dh) - |t_i - t_j|^2, softmax, AV.
// grid = (N/16, H, B), block = 256 (8 waves). WMMA for QK^T and AV.
// Frame translations staged into LDS with gfx1250 async global->LDS loads.
// ---------------------------------------------------------------------------
__global__ void attn_kernel(const float* __restrict__ Q,
                            const float* __restrict__ K,
                            const float* __restrict__ V,
                            const float* __restrict__ Fr,
                            float* __restrict__ O) {
    __shared__ float sS[16 * NN];      // scores -> probs
    __shared__ float sTj[NN * 3];
    __shared__ float sTi[16 * 3];

    int tid  = threadIdx.x;
    int lane = tid & 31;
    int wave = tid >> 5;
    int i0 = blockIdx.x * 16;
    int h  = blockIdx.y;
    int b  = blockIdx.z;

    // Stage t_j (frame translations) into LDS via async global->LDS (ASYNCcnt).
    for (int e = tid; e < NN * 3; e += 256) {
        int j = e / 3, c = e % 3;                       // t[c] = frame[c*4+3]
        unsigned lds_off = (unsigned)(size_t)&sTj[e];
        unsigned long long ga =
            (unsigned long long)(Fr + (size_t)(b * NN + j) * 16 + (c * 4 + 3));
        asm volatile("global_load_async_to_lds_b32 %0, %1, off"
                     :: "v"(lds_off), "v"(ga) : "memory");
    }
    if (tid < 48) {
        int r = tid / 3, c = tid % 3;
        sTi[tid] = Fr[(size_t)(b * NN + i0 + r) * 16 + (c * 4 + 3)];
    }
    asm volatile("s_wait_asynccnt 0x0" ::: "memory");
    __syncthreads();

    int m = lane & 15;
    int n = lane & 15;

    // Q fragment for this 16-row block (Dh = 32 -> single K-step)
    float qf[16];
    load_a_frag(Q + ((size_t)(b * NN + i0 + m) * DD + h * DH), lane, qf);
    v16bf qh, ql;
    cvt_split(qf, qh, ql);

    const float scale = 0.17677669529663687f;  // 1/sqrt(32)

    // 24 j-tiles across 8 waves (3 each)
#pragma unroll
    for (int tt = 0; tt < 3; ++tt) {
        int jt = wave * 3 + tt;
        float kf[16];
        // K^T fragment: element j of the frag indexes d (contiguous)
        load_b_frag(K + ((size_t)(b * NN + jt * 16 + n) * DD + h * DH), 1, lane, kf);
        v16bf kh, kl;
        cvt_split(kf, kh, kl);
        v8f acc = {};
        acc = wmma_split3(qh, ql, kh, kl, acc);
#pragma unroll
        for (int r = 0; r < 8; ++r) {
            int il = r + ((lane < 16) ? 0 : 8);
            int j  = jt * 16 + n;
            float dx = sTi[il * 3 + 0] - sTj[j * 3 + 0];
            float dy = sTi[il * 3 + 1] - sTj[j * 3 + 1];
            float dz = sTi[il * 3 + 2] - sTj[j * 3 + 2];
            sS[il * NN + j] = acc[r] * scale - (dx * dx + dy * dy + dz * dz);
        }
    }
    __syncthreads();

    // softmax over j (each wave owns 2 rows)
#pragma unroll
    for (int rr = 0; rr < 2; ++rr) {
        int row = wave * 2 + rr;
        float mx = -1e30f;
        for (int j = lane; j < NN; j += 32) mx = fmaxf(mx, sS[row * NN + j]);
        mx = wave_max(mx);
        float sum = 0.f;
        for (int j = lane; j < NN; j += 32) {
            float e = expf(sS[row * NN + j] - mx);
            sS[row * NN + j] = e;
            sum += e;
        }
        sum = wave_sum(sum);
        float inv = 1.f / sum;
        for (int j = lane; j < NN; j += 32) sS[row * NN + j] *= inv;
    }
    __syncthreads();

    // AV: two 16x16 d-tiles, waves 0 and 1
    if (wave < 2) {
        int dt = wave;
        v8f acc = {};
        for (int j0 = 0; j0 < NN; j0 += 32) {
            float aa[16], bb[16];
            load_a_frag(&sS[m * NN + j0], lane, aa);
            load_b_frag(V + ((size_t)(b * NN + j0) * DD + h * DH + dt * 16 + n), DD, lane, bb);
            v16bf ah, al, vh, vl;
            cvt_split(aa, ah, al);
            cvt_split(bb, vh, vl);
            acc = wmma_split3(ah, al, vh, vl, acc);
        }
#pragma unroll
        for (int r = 0; r < 8; ++r) {
            int row = i0 + r + ((lane < 16) ? 0 : 8);
            O[(size_t)(b * NN + row) * DD + h * DH + dt * 16 + n] = acc[r];
        }
    }
}

// ---------------------------------------------------------------------------
// Embedding + sinusoidal positional encoding (pre-LN), block = D
// ---------------------------------------------------------------------------
__global__ void embed_kernel(const int* __restrict__ tokens,
                             const float* __restrict__ tok_emb,
                             float* __restrict__ out) {
    int row = blockIdx.x;          // 0..MM-1
    int d   = threadIdx.x;         // 0..DD-1
    int tok = tokens[row];
    int nn  = row % NN;
    int i2  = d >> 1;
    float div = expf(-(float)(2 * i2) * (logf(10000.f) / (float)DD));
    float ang = (float)nn * div;
    float pe  = (d & 1) ? cosf(ang) : sinf(ang);
    out[(size_t)row * DD + d] = tok_emb[(size_t)tok * DD + d] + pe;
}

// ---------------------------------------------------------------------------
// LayerNorm over last dim D = blockDim.x (256); population variance, eps=1e-5
// ---------------------------------------------------------------------------
__global__ void layernorm_kernel(const float* __restrict__ in,
                                 const float* __restrict__ w,
                                 const float* __restrict__ b,
                                 float* __restrict__ out) {
    __shared__ float red[DD];
    int row = blockIdx.x;
    int t   = threadIdx.x;
    float v = in[(size_t)row * DD + t];
    red[t] = v;
    __syncthreads();
#pragma unroll
    for (int s = DD / 2; s > 0; s >>= 1) {
        if (t < s) red[t] += red[t + s];
        __syncthreads();
    }
    float mean = red[0] * (1.f / DD);
    __syncthreads();
    float e = v - mean;
    red[t] = e * e;
    __syncthreads();
#pragma unroll
    for (int s = DD / 2; s > 0; s >>= 1) {
        if (t < s) red[t] += red[t + s];
        __syncthreads();
    }
    float var = red[0] * (1.f / DD);
    out[(size_t)row * DD + t] = e * rsqrtf(var + 1e-5f) * w[t] + b[t];
}

// ---------------------------------------------------------------------------
// pairmean[b,i,:] = (1/N) * sum_j LN64(proj[b,i] + proj[b,j] + rel_e[clip(j-i)+64])
// one wave per i; each lane owns 2 of the 64 dims; shuffle reductions for LN.
// ---------------------------------------------------------------------------
__global__ void pairmean_kernel(const float* __restrict__ proj,
                                const float* __restrict__ relpos,
                                const float* __restrict__ plw,
                                const float* __restrict__ plb,
                                float* __restrict__ pairmean) {
    int lane = threadIdx.x & 31;
    int wave = threadIdx.x >> 5;
    int gi = blockIdx.x * 8 + wave;       // 0..MM-1
    int b  = gi / NN;
    int i  = gi % NN;
    int d0 = lane * 2, d1 = d0 + 1;

    float pi0 = proj[(size_t)gi * PD + d0];
    float pi1 = proj[(size_t)gi * PD + d1];
    float w0 = plw[d0], w1 = plw[d1];
    float b0 = plb[d0], b1 = plb[d1];

    float acc0 = 0.f, acc1 = 0.f;
    for (int j = 0; j < NN; ++j) {
        int rel = j - i;
        rel = (rel < -64) ? -64 : (rel > 64 ? 64 : rel);
        rel += 64;
        const float* re = relpos + (size_t)rel * PD;
        const float* pj = proj + (size_t)(b * NN + j) * PD;
        float v0 = pi0 + pj[d0] + re[d0];
        float v1 = pi1 + pj[d1] + re[d1];
        float mean = wave_sum(v0 + v1) * (1.f / 64.f);
        float e0 = v0 - mean, e1 = v1 - mean;
        float var = wave_sum(e0 * e0 + e1 * e1) * (1.f / 64.f);
        float r = rsqrtf(var + 1e-5f);
        acc0 += e0 * r * w0 + b0;
        acc1 += e1 * r * w1 + b1;
    }
    pairmean[(size_t)gi * PD + d0] = acc0 * (1.f / NN);
    pairmean[(size_t)gi * PD + d1] = acc1 * (1.f / NN);
}

// ---------------------------------------------------------------------------
// Frame head: feat = [x(256) | pairmean(64)]; h = silu(feat@Wf1+bf1) (128);
// raw = h@Wf2+bf2 (9); rot6d -> R; frames[b,i] = [[R | t],[0 0 0 1]]
// one wave per (b,i), 8 waves per block.
// ---------------------------------------------------------------------------
__global__ void frames_kernel(const float* __restrict__ x,
                              const float* __restrict__ pairmean,
                              const float* __restrict__ Wf1,
                              const float* __restrict__ bf1,
                              const float* __restrict__ Wf2,
                              const float* __restrict__ bf2,
                              float* __restrict__ frames) {
    __shared__ float hsh[8][128];
    __shared__ float rsh[8][9];
    int lane = threadIdx.x & 31;
    int wave = threadIdx.x >> 5;
    int gi = blockIdx.x * 8 + wave;

    const float* xr = x + (size_t)gi * DD;
    const float* pm = pairmean + (size_t)gi * PD;

#pragma unroll
    for (int q = 0; q < 4; ++q) {
        int o = lane * 4 + q;
        float a = bf1[o];
        for (int t = 0; t < DD; ++t) a += xr[t] * Wf1[(size_t)t * 128 + o];
        for (int t = 0; t < PD; ++t) a += pm[t] * Wf1[(size_t)(DD + t) * 128 + o];
        hsh[wave][o] = a / (1.f + expf(-a));   // SiLU
    }
    __syncthreads();
    if (lane < 9) {
        float a = bf2[lane];
        for (int t = 0; t < 128; ++t) a += hsh[wave][t] * Wf2[(size_t)t * 9 + lane];
        rsh[wave][lane] = a;
    }
    __syncthreads();
    if (lane == 0) {
        const float* r = rsh[wave];
        float a1x = r[0], a1y = r[1], a1z = r[2];
        float a2x = r[3], a2y = r[4], a2z = r[5];
        float n1 = sqrtf(a1x * a1x + a1y * a1y + a1z * a1z + 1e-8f);
        float b1x = a1x / n1, b1y = a1y / n1, b1z = a1z / n1;
        float d = b1x * a2x + b1y * a2y + b1z * a2z;
        float px = a2x - d * b1x, py = a2y - d * b1y, pz = a2z - d * b1z;
        float n2 = sqrtf(px * px + py * py + pz * pz + 1e-8f);
        float b2x = px / n2, b2y = py / n2, b2z = pz / n2;
        float b3x = b1y * b2z - b1z * b2y;
        float b3y = b1z * b2x - b1x * b2z;
        float b3z = b1x * b2y - b1y * b2x;
        float* F = frames + (size_t)gi * 16;
        F[0] = b1x;  F[1] = b1y;  F[2]  = b1z;  F[3]  = r[6];
        F[4] = b2x;  F[5] = b2y;  F[6]  = b2z;  F[7]  = r[7];
        F[8] = b3x;  F[9] = b3y;  F[10] = b3z;  F[11] = r[8];
        F[12] = 0.f; F[13] = 0.f; F[14] = 0.f;  F[15] = 1.f;
    }
}

__global__ void init_frames_kernel(float* __restrict__ frames) {
    int idx = blockIdx.x * blockDim.x + threadIdx.x;
    if (idx >= MM * 16) return;
    int r = (idx >> 2) & 3;
    int c = idx & 3;
    frames[idx] = (r == c) ? 1.f : 0.f;
}

__global__ void copyout_kernel(const float* __restrict__ frames,
                               const float* __restrict__ x,
                               float* __restrict__ out, int total) {
    int i = blockIdx.x * blockDim.x + threadIdx.x;
    if (i >= total) return;
    out[i] = (i < MM * 16) ? frames[i] : x[i - MM * 16];
}

// ---------------------------------------------------------------------------
// Host-side orchestration
// ---------------------------------------------------------------------------
extern "C" void kernel_launch(void* const* d_in, const int* in_sizes, int n_in,
                              void* d_out, int out_size, void* d_ws, size_t ws_size,
                              hipStream_t stream) {
    (void)in_sizes; (void)n_in; (void)ws_size;

    const int*   tokens     = (const int*)  d_in[0];
    const float* tok_emb    = (const float*)d_in[1];
    const float* emb_ln_w   = (const float*)d_in[2];
    const float* emb_ln_b   = (const float*)d_in[3];
    const float* Wq         = (const float*)d_in[4];
    const float* Wk         = (const float*)d_in[5];
    const float* Wv         = (const float*)d_in[6];
    const float* Wo         = (const float*)d_in[7];
    const float* bo         = (const float*)d_in[8];
    const float* attn_ln_w  = (const float*)d_in[9];
    const float* attn_ln_b  = (const float*)d_in[10];
    const float* ff_ln_w    = (const float*)d_in[11];
    const float* ff_ln_b    = (const float*)d_in[12];
    const float* W1         = (const float*)d_in[13];
    const float* b1         = (const float*)d_in[14];
    const float* W2         = (const float*)d_in[15];
    const float* b2         = (const float*)d_in[16];
    const float* Wop        = (const float*)d_in[17];
    const float* bop        = (const float*)d_in[18];
    const float* relpos_emb = (const float*)d_in[19];
    const float* pair_ln_w  = (const float*)d_in[20];
    const float* pair_ln_b  = (const float*)d_in[21];
    const float* Wf1        = (const float*)d_in[22];
    const float* bf1        = (const float*)d_in[23];
    const float* Wf2        = (const float*)d_in[24];
    const float* bf2        = (const float*)d_in[25];

    // workspace carve-out (256B aligned chunks)
    char* wsb = (char*)d_ws;
    auto carve = [&](size_t bytes) -> void* {
        void* p = (void*)wsb;
        wsb += (bytes + 255) & ~(size_t)255;
        return p;
    };
    float* x      = (float*)carve((size_t)MM * DD * 4);
    float* tmp    = (float*)carve((size_t)MM * DD * 4);
    float* q      = (float*)carve((size_t)MM * DD * 4);
    float* kbuf   = (float*)carve((size_t)MM * DD * 4);
    float* v      = (float*)carve((size_t)MM * DD * 4);
    float* attno  = (float*)carve((size_t)MM * DD * 4);
    float* h1     = (float*)carve((size_t)MM * FF * 4);
    float* proj   = (float*)carve((size_t)MM * PD * 4);
    float* pairm  = (float*)carve((size_t)MM * PD * 4);
    float* frames = (float*)carve((size_t)MM * 16 * 4);
    __bf16* Ah  = (__bf16*)carve((size_t)MM * FF * 2);   // max A: h1 (768x1024)
    __bf16* Al  = (__bf16*)carve((size_t)MM * FF * 2);
    __bf16* Wth = (__bf16*)carve((size_t)FF * DD * 2);   // max W: 1024x256 / 256x1024
    __bf16* Wtl = (__bf16*)carve((size_t)FF * DD * 2);

    // embed + LN
    embed_kernel<<<MM, DD, 0, stream>>>(tokens, tok_emb, tmp);
    layernorm_kernel<<<MM, DD, 0, stream>>>(tmp, emb_ln_w, emb_ln_b, x);
    init_frames_kernel<<<(MM * 16 + 255) / 256, 256, 0, stream>>>(frames);

    dim3 gD(MM / 64, DD / 64);    // N=256: (12,4)
    dim3 gF(MM / 64, FF / 64);    // N=1024: (12,16)
    dim3 gP(MM / 64, 1);          // N=64:  (12,1)
    dim3 gAttn(NN / 16, HH, BB);

    const int CT = 256;
    int nAD  = MM * DD,  gAD  = (nAD  + CT - 1) / CT;   // activation DxM
    int nAF  = MM * FF,  gAF  = (nAF  + CT - 1) / CT;   // activation FFxM
    int nWDD = DD * DD,  gWDD = (nWDD + CT - 1) / CT;
    int nWDF = DD * FF,  gWDF = (nWDF + CT - 1) / CT;
    int nWDP = DD * PD,  gWDP = (nWDP + CT - 1) / CT;

    for (int l = 0; l < LL; ++l) {
        const float* Wq_l = Wq + (size_t)l * DD * DD;
        const float* Wk_l = Wk + (size_t)l * DD * DD;
        const float* Wv_l = Wv + (size_t)l * DD * DD;
        const float* Wo_l = Wo + (size_t)l * DD * DD;
        const float* bo_l = bo + (size_t)l * DD;
        const float* W1_l = W1 + (size_t)l * DD * FF;
        const float* b1_l = b1 + (size_t)l * FF;
        const float* W2_l = W2 + (size_t)l * FF * DD;
        const float* b2_l = b2 + (size_t)l * DD;

        // Q/K/V projections: convert x once, reuse for all three GEMMs
        cvt_act_kernel<<<gAD, CT, 0, stream>>>(x, Ah, Al, nAD);
        cvt_wt_kernel<<<gWDD, CT, 0, stream>>>(Wq_l, Wth, Wtl, DD, DD);
        gemm_wmma_kernel<0, false><<<gD, 128, 0, stream>>>(Ah, Al, Wth, Wtl, nullptr, nullptr, q,    MM, DD, DD);
        cvt_wt_kernel<<<gWDD, CT, 0, stream>>>(Wk_l, Wth, Wtl, DD, DD);
        gemm_wmma_kernel<0, false><<<gD, 128, 0, stream>>>(Ah, Al, Wth, Wtl, nullptr, nullptr, kbuf, MM, DD, DD);
        cvt_wt_kernel<<<gWDD, CT, 0, stream>>>(Wv_l, Wth, Wtl, DD, DD);
        gemm_wmma_kernel<0, false><<<gD, 128, 0, stream>>>(Ah, Al, Wth, Wtl, nullptr, nullptr, v,    MM, DD, DD);

        // SE(3) frame attention
        attn_kernel<<<gAttn, 256, 0, stream>>>(q, kbuf, v, frames, attno);

        // out proj + residual, then LN -> x
        cvt_act_kernel<<<gAD, CT, 0, stream>>>(attno, Ah, Al, nAD);
        cvt_wt_kernel<<<gWDD, CT, 0, stream>>>(Wo_l, Wth, Wtl, DD, DD);
        gemm_wmma_kernel<0, true><<<gD, 128, 0, stream>>>(Ah, Al, Wth, Wtl, bo_l, x, tmp, MM, DD, DD);
        layernorm_kernel<<<MM, DD, 0, stream>>>(tmp, attn_ln_w + (size_t)l * DD, attn_ln_b + (size_t)l * DD, x);

        // FFN: h = LN(x); x = x + gelu(h@W1+b1)@W2 + b2
        layernorm_kernel<<<MM, DD, 0, stream>>>(x, ff_ln_w + (size_t)l * DD, ff_ln_b + (size_t)l * DD, tmp);
        cvt_act_kernel<<<gAD, CT, 0, stream>>>(tmp, Ah, Al, nAD);
        cvt_wt_kernel<<<gWDF, CT, 0, stream>>>(W1_l, Wth, Wtl, DD, FF);
        gemm_wmma_kernel<1, false><<<gF, 128, 0, stream>>>(Ah, Al, Wth, Wtl, b1_l, nullptr, h1, MM, DD, FF);
        cvt_act_kernel<<<gAF, CT, 0, stream>>>(h1, Ah, Al, nAF);
        cvt_wt_kernel<<<gWDF, CT, 0, stream>>>(W2_l, Wth, Wtl, FF, DD);
        gemm_wmma_kernel<0, true ><<<gD, 128, 0, stream>>>(Ah, Al, Wth, Wtl, b2_l, x, x, MM, FF, DD);

        // pair branch
        cvt_act_kernel<<<gAD, CT, 0, stream>>>(x, Ah, Al, nAD);
        cvt_wt_kernel<<<gWDP, CT, 0, stream>>>(Wop, Wth, Wtl, DD, PD);
        gemm_wmma_kernel<2, false><<<gP, 128, 0, stream>>>(Ah, Al, Wth, Wtl, bop, nullptr, proj, MM, DD, PD);
        pairmean_kernel<<<MM / 8, 256, 0, stream>>>(proj, relpos_emb, pair_ln_w, pair_ln_b, pairm);

        // frame head -> frames (used by next layer's attention)
        frames_kernel<<<MM / 8, 256, 0, stream>>>(x, pairm, Wf1, bf1, Wf2, bf2, frames);
    }

    copyout_kernel<<<(out_size + 255) / 256, 256, 0, stream>>>(frames, x, (float*)d_out, out_size);
}
